// InsiderClassifier_84748294685114
// MI455X (gfx1250) — matrix operand order repaired and checked by reading, not compiled, not run on hardware
//
#include <hip/hip_runtime.h>
#include <hip/hip_bf16.h>
#include <math.h>

#define BB 32
#define NN 512
#define DD 256
#define HH 256
#define CC 2

typedef __attribute__((ext_vector_type(16))) __bf16       v16bf;
typedef __attribute__((ext_vector_type(8)))  float        v8f;
typedef __attribute__((ext_vector_type(4)))  unsigned int v4u;

union Frag16 { v16bf v; v4u q[2]; };

__device__ __forceinline__ unsigned short f2bf(float f) {
    unsigned int u = __float_as_uint(f);
    unsigned int r = (u + 0x7FFFu + ((u >> 16) & 1u)) >> 16;
    return (unsigned short)r;
}

// A fragment (16x32 bf16, row-major source, ld in elements).
// lanes 0-15: M=ml, v0..v3 = K 0..7, v4..v7 = K 16..23
// lanes 16-31: M=ml, v0..v3 = K 8..15, v4..v7 = K 24..31
__device__ __forceinline__ void load_a(Frag16& f, const unsigned short* A, int ld,
                                       int m0, int k0, int ml, int lg) {
    const unsigned short* p = A + (size_t)(m0 + ml) * ld + k0 + lg * 8;
    f.q[0] = *(const v4u*)p;
    f.q[1] = *(const v4u*)(p + 16);
}

// B fragment (32x16 bf16) loaded from the TRANSPOSED matrix BT[n][k] (row-major).
// lanes 0-15: N=ml, K = k0..k0+7, k0+16..k0+23 ; lanes 16-31: N=ml, K = k0+8..k0+15, k0+24..k0+31
__device__ __forceinline__ void load_bt(Frag16& f, const unsigned short* BT, int ld,
                                        int n0, int k0, int ml, int lg) {
    const unsigned short* p = BT + (size_t)(n0 + ml) * ld + k0 + lg * 8;
    f.q[0] = *(const v4u*)p;
    f.q[1] = *(const v4u*)(p + 16);
}

// 16x64 output block per wave: one shared A fragment, 4 B fragments, 4 WMMAs per
// K-step, with explicit double-buffering so step k+1's loads are in flight while
// step k's WMMAs execute.
__device__ __forceinline__ void wmma_tile4(v8f acc[4],
                                           const unsigned short* A, int lda,
                                           const unsigned short* BT, int ldb,
                                           int m0, int n0, int K, int ml, int lg) {
    Frag16 fa0, fb0[4];
    load_a(fa0, A, lda, m0, 0, ml, lg);
#pragma unroll
    for (int j = 0; j < 4; ++j) load_bt(fb0[j], BT, ldb, n0 + j * 16, 0, ml, lg);

#pragma unroll
    for (int k0 = 0; k0 < K; k0 += 32) {
        Frag16 fa1, fb1[4];
        int kn = k0 + 32;
        if (kn < K) {
            load_a(fa1, A, lda, m0, kn, ml, lg);
#pragma unroll
            for (int j = 0; j < 4; ++j) load_bt(fb1[j], BT, ldb, n0 + j * 16, kn, ml, lg);
            // k+2 slice to WGP-scope caches (locality 3 -> near prefetch)
            if (kn + 32 < K)
                __builtin_prefetch(A + (size_t)(m0 + ml) * lda + kn + 32, 0, 3);
        }
#pragma unroll
        for (int j = 0; j < 4; ++j)
            acc[j] = __builtin_amdgcn_wmma_f32_16x16x32_bf16(
                false, fa0.v, false, fb0[j].v, (short)0, acc[j], false, false);
        fa0 = fa1;
#pragma unroll
        for (int j = 0; j < 4; ++j) fb0[j] = fb1[j];
    }
}

// ---- 1. normalize rows, emit bf16 cn and bf16 context-transpose, store |x|^2 ----
__global__ __launch_bounds__(256) void k_normalize(const float* __restrict__ ctx,
        unsigned short* __restrict__ cnb, unsigned short* __restrict__ ctxT,
        float* __restrict__ sqnorm) {
    int wid = threadIdx.x >> 5, lane = threadIdx.x & 31;
    int row = blockIdx.x * 8 + wid;                 // b*N + n
    const float* x = ctx + (size_t)row * DD;
    float v[8]; float ss = 0.f;
    for (int i = 0; i < 8; ++i) { v[i] = x[lane + i * 32]; ss += v[i] * v[i]; }
    for (int off = 16; off >= 1; off >>= 1) ss += __shfl_xor(ss, off, 32);
    float inv = 1.0f / fmaxf(sqrtf(ss), 1e-12f);
    unsigned short* c = cnb + (size_t)row * DD;
    for (int i = 0; i < 8; ++i) c[lane + i * 32] = f2bf(v[i] * inv);
    int b = row / NN, n = row % NN;
    unsigned short* t = ctxT + (size_t)b * DD * NN;
    for (int i = 0; i < 8; ++i) t[(size_t)(lane + i * 32) * NN + n] = f2bf(v[i]);
    if (lane == 0) sqnorm[row] = ss;
}

// ---- weight transpose + bf16 convert (WT[c][r] = W[r][c]) ----
__global__ __launch_bounds__(256) void k_transpose_w(const float* __restrict__ W,
        unsigned short* __restrict__ WT, int rows, int cols) {
    int i = blockIdx.x * 256 + threadIdx.x;
    if (i >= rows * cols) return;
    int r = i / cols, c = i % cols;
    WT[(size_t)c * rows + r] = f2bf(W[i]);
}

// ---- 2. S = cn . cn^T  (WMMA), eps-threshold, rowsum + sum(adj^2) ----
// one wave -> 16x64 tile; grid = BB * 32 * 8
__global__ __launch_bounds__(32) void k_similarity(const unsigned short* __restrict__ cnb,
        unsigned short* __restrict__ adjb, float* __restrict__ rowsum,
        float* __restrict__ sadj2) {
    int tile = blockIdx.x;
    int b = tile >> 8, tm = (tile >> 3) & 31, tn = tile & 7;
    int lane = threadIdx.x & 31, ml = lane & 15, lg = lane >> 4;
    const unsigned short* base = cnb + (size_t)b * NN * DD;
    v8f acc[4] = {};
    wmma_tile4(acc, base, DD, base, DD, tm * 16, tn * 64, DD, ml, lg);
    unsigned short* arow = adjb + (size_t)b * NN * NN;
    float pr[8] = {0, 0, 0, 0, 0, 0, 0, 0};
    float a2 = 0.f;
#pragma unroll
    for (int j = 0; j < 4; ++j) {
        int n = tn * 64 + j * 16 + ml;
#pragma unroll
        for (int r = 0; r < 8; ++r) {
            float s = acc[j][r];
            float v = (s > 0.3f) ? s : 0.0f;
            int m = tm * 16 + lg * 8 + r;
            arow[(size_t)m * NN + n] = f2bf(v);
            pr[r] += v; a2 += v * v;
        }
    }
    for (int off = 8; off >= 1; off >>= 1)
        for (int r = 0; r < 8; ++r) pr[r] += __shfl_xor(pr[r], off, 32);
    if (ml == 0)
        for (int r = 0; r < 8; ++r)
            atomicAdd(&rowsum[b * NN + tm * 16 + lg * 8 + r], pr[r]);
    for (int off = 16; off >= 1; off >>= 1) a2 += __shfl_xor(a2, off, 32);
    if (lane == 0) atomicAdd(sadj2, a2);
}

// ---- 3. M = adj . context (WMMA); fuse smooth-loss dot and row-normalized bf16 out ----
// grid = BB * 32 * 4
__global__ __launch_bounds__(32) void k_diffuse(const unsigned short* __restrict__ adjb,
        const unsigned short* __restrict__ ctxT, const float* __restrict__ ctx,
        const float* __restrict__ rowsum, unsigned short* __restrict__ hinb,
        float* __restrict__ sdot) {
    int tile = blockIdx.x;
    int b = tile >> 7, tm = (tile >> 2) & 31, tn = tile & 3;
    int lane = threadIdx.x & 31, ml = lane & 15, lg = lane >> 4;
    const unsigned short* A  = adjb + (size_t)b * NN * NN;
    const unsigned short* BT = ctxT + (size_t)b * DD * NN;
    v8f acc[4] = {};
    wmma_tile4(acc, A, NN, BT, NN, tm * 16, tn * 64, NN, ml, lg);
    float dot = 0.f;
#pragma unroll
    for (int j = 0; j < 4; ++j) {
        int d = tn * 64 + j * 16 + ml;
#pragma unroll
        for (int r = 0; r < 8; ++r) {
            int m = tm * 16 + lg * 8 + r;
            float Mv = acc[j][r];
            dot += ctx[((size_t)b * NN + m) * DD + d] * Mv;
            float rw = rowsum[b * NN + m] + 1e-12f;
            hinb[((size_t)b * NN + m) * DD + d] = f2bf(Mv / rw);
        }
    }
    for (int off = 16; off >= 1; off >>= 1) dot += __shfl_xor(dot, off, 32);
    if (lane == 0) atomicAdd(sdot, dot);
}

// ---- 4. h = relu(hin @ W_gcn + b) (WMMA); grid = BB * 32 * 4 ----
__global__ __launch_bounds__(32) void k_gcn(const unsigned short* __restrict__ hinb,
        const unsigned short* __restrict__ WgcnT, const float* __restrict__ bgcn,
        float* __restrict__ hbuf, unsigned short* __restrict__ hb) {
    int tile = blockIdx.x;
    int b = tile >> 7, tm = (tile >> 2) & 31, tn = tile & 3;
    int lane = threadIdx.x & 31, ml = lane & 15, lg = lane >> 4;
    const unsigned short* A = hinb + (size_t)b * NN * DD;
    v8f acc[4] = {};
    wmma_tile4(acc, A, DD, WgcnT, DD, tm * 16, tn * 64, DD, ml, lg);
#pragma unroll
    for (int j = 0; j < 4; ++j) {
        int hcol = tn * 64 + j * 16 + ml;
        float bias = bgcn[hcol];
#pragma unroll
        for (int r = 0; r < 8; ++r) {
            int m = tm * 16 + lg * 8 + r;
            float v = fmaxf(acc[j][r] + bias, 0.0f);
            size_t idx = ((size_t)b * NN + m) * HH + hcol;
            hbuf[idx] = v;
            hb[idx] = f2bf(v);
        }
    }
}

// ---- 5. a = h @ W_att1 + b (WMMA); grid = BB * 32 * 4 ----
__global__ __launch_bounds__(32) void k_att1(const unsigned short* __restrict__ hb,
        const unsigned short* __restrict__ Wat1T, const float* __restrict__ batt1,
        float* __restrict__ abuf) {
    int tile = blockIdx.x;
    int b = tile >> 7, tm = (tile >> 2) & 31, tn = tile & 3;
    int lane = threadIdx.x & 31, ml = lane & 15, lg = lane >> 4;
    const unsigned short* A = hb + (size_t)b * NN * HH;
    v8f acc[4] = {};
    wmma_tile4(acc, A, HH, Wat1T, HH, tm * 16, tn * 64, HH, ml, lg);
#pragma unroll
    for (int j = 0; j < 4; ++j) {
        int hcol = tn * 64 + j * 16 + ml;
        float bias = batt1[hcol];
#pragma unroll
        for (int r = 0; r < 8; ++r) {
            int m = tm * 16 + lg * 8 + r;
            abuf[((size_t)b * NN + m) * HH + hcol] = acc[j][r] + bias;
        }
    }
}

// ---- 6. LayerNorm -> exact GELU -> dot(W_att2) -> masked logit; one wave per row ----
__global__ __launch_bounds__(256) void k_lngelu(const float* __restrict__ abuf,
        const float* __restrict__ lng, const float* __restrict__ lnb,
        const float* __restrict__ Watt2, const float* __restrict__ batt2,
        const int* __restrict__ amask, float* __restrict__ wlogit) {
    int wid = threadIdx.x >> 5, lane = threadIdx.x & 31;
    int row = blockIdx.x * 8 + wid;
    const float* a = abuf + (size_t)row * HH;
    float x[8]; float s = 0.f, s2 = 0.f;
    for (int i = 0; i < 8; ++i) { x[i] = a[lane + i * 32]; s += x[i]; s2 += x[i] * x[i]; }
    for (int off = 16; off >= 1; off >>= 1) {
        s  += __shfl_xor(s,  off, 32);
        s2 += __shfl_xor(s2, off, 32);
    }
    float mu   = s * (1.0f / HH);
    float var  = s2 * (1.0f / HH) - mu * mu;
    float rstd = rsqrtf(var + 1e-5f);
    float dot = 0.f;
    for (int i = 0; i < 8; ++i) {
        int hcol = lane + i * 32;
        float y = (x[i] - mu) * rstd * lng[hcol] + lnb[hcol];
        float g = 0.5f * y * (1.0f + erff(y * 0.70710678118654752f));
        dot += g * Watt2[hcol];
    }
    for (int off = 16; off >= 1; off >>= 1) dot += __shfl_xor(dot, off, 32);
    if (lane == 0) {
        float w = dot + batt2[0];
        if (amask[row] == 0) w = -__builtin_inff();
        wlogit[row] = w;
    }
}

// ---- 7. softmax over N, pooled = sum w*h, logits = pooled @ W_out + b ----
__global__ __launch_bounds__(256) void k_pool(const float* __restrict__ wlogit,
        const float* __restrict__ hbuf, const float* __restrict__ Wout,
        const float* __restrict__ bout, float* __restrict__ out) {
    __shared__ float wl[NN];
    __shared__ float red[256];
    int b = blockIdx.x, t = threadIdx.x;
    float m = -__builtin_inff();
    for (int n = t; n < NN; n += 256) { float v = wlogit[b * NN + n]; wl[n] = v; m = fmaxf(m, v); }
    red[t] = m; __syncthreads();
    for (int off = 128; off >= 1; off >>= 1) {
        if (t < off) red[t] = fmaxf(red[t], red[t + off]);
        __syncthreads();
    }
    float mx = red[0]; __syncthreads();
    float ls = 0.f;
    for (int n = t; n < NN; n += 256) { float e = expf(wl[n] - mx); wl[n] = e; ls += e; }
    red[t] = ls; __syncthreads();
    for (int off = 128; off >= 1; off >>= 1) {
        if (t < off) red[t] += red[t + off];
        __syncthreads();
    }
    float inv = 1.0f / red[0]; __syncthreads();
    float acc = 0.f;
    const float* hp = hbuf + (size_t)b * NN * HH + t;
    for (int n = 0; n < NN; ++n) acc += wl[n] * hp[(size_t)n * HH];
    red[t] = acc * inv; __syncthreads();
    if (t < CC) {
        float lg = bout[t];
        for (int hcol = 0; hcol < HH; ++hcol) lg += red[hcol] * Wout[hcol * CC + t];
        out[b * CC + t] = lg;
    }
}

// ---- 8. sum(row * |ctx_row|^2) for smooth loss ----
__global__ __launch_bounds__(256) void k_rowq(const float* __restrict__ rowsum,
        const float* __restrict__ sqnorm, float* __restrict__ srowq) {
    __shared__ float red[256];
    int i = blockIdx.x * 256 + threadIdx.x;
    red[threadIdx.x] = rowsum[i] * sqnorm[i];
    __syncthreads();
    for (int off = 128; off >= 1; off >>= 1) {
        if (threadIdx.x < off) red[threadIdx.x] += red[threadIdx.x + off];
        __syncthreads();
    }
    if (threadIdx.x == 0) atomicAdd(srowq, red[0]);
}

// ---- 9. assemble graph loss ----
__global__ void k_finalize(const float* __restrict__ scal, float* __restrict__ out) {
    // scal: [0]=sum(ctx.M), [1]=sum(row*|ctx|^2), [2]=sum(adj^2); DEGREE_R == 0
    float smooth_sum = scal[1] - scal[0];
    out[BB * CC] = (0.2f * smooth_sum + 0.1f * scal[2]) / (float)(BB * NN * NN);
}

extern "C" void kernel_launch(void* const* d_in, const int* in_sizes, int n_in,
                              void* d_out, int out_size, void* d_ws, size_t ws_size,
                              hipStream_t stream) {
    (void)in_sizes; (void)n_in; (void)out_size; (void)ws_size;
    const float* ctx   = (const float*)d_in[0];
    const int*   amask = (const int*)d_in[1];
    const float* Wgcn  = (const float*)d_in[2];
    const float* bgcn  = (const float*)d_in[3];
    const float* Watt1 = (const float*)d_in[4];
    const float* batt1 = (const float*)d_in[5];
    const float* lng   = (const float*)d_in[6];
    const float* lnb   = (const float*)d_in[7];
    const float* Watt2 = (const float*)d_in[8];
    const float* batt2 = (const float*)d_in[9];
    const float* Wout  = (const float*)d_in[10];
    const float* bout  = (const float*)d_in[11];
    float* out = (float*)d_out;

    char* ws = (char*)d_ws;
    size_t off = 0;
    auto alloc = [&](size_t bytes) -> char* {
        char* p = ws + off;
        off += (bytes + 255) & ~(size_t)255;
        return p;
    };
    unsigned short* cnb   = (unsigned short*)alloc((size_t)BB * NN * DD * 2);
    unsigned short* ctxT  = (unsigned short*)alloc((size_t)BB * DD * NN * 2);
    unsigned short* adjb  = (unsigned short*)alloc((size_t)BB * NN * NN * 2);
    unsigned short* hinb  = (unsigned short*)alloc((size_t)BB * NN * DD * 2);
    unsigned short* hb    = (unsigned short*)alloc((size_t)BB * NN * HH * 2);
    unsigned short* WgcnT = (unsigned short*)alloc((size_t)DD * HH * 2);
    unsigned short* Wat1T = (unsigned short*)alloc((size_t)HH * HH * 2);
    float* abuf   = (float*)alloc((size_t)BB * NN * HH * 4);
    float* hbuf   = (float*)alloc((size_t)BB * NN * HH * 4);
    float* sqnorm = (float*)alloc((size_t)BB * NN * 4);
    float* wlog   = (float*)alloc((size_t)BB * NN * 4);
    float* rowsum = (float*)alloc((size_t)BB * NN * 4);
    float* scal   = (float*)alloc(4 * sizeof(float));   // [sdot, srowq, sadj2]

    hipMemsetAsync(rowsum, 0, (size_t)BB * NN * 4, stream);
    hipMemsetAsync(scal, 0, 4 * sizeof(float), stream);

    k_normalize  <<<BB * NN / 8, 256, 0, stream>>>(ctx, cnb, ctxT, sqnorm);
    k_transpose_w<<<(DD * HH + 255) / 256, 256, 0, stream>>>(Wgcn,  WgcnT, DD, HH);
    k_transpose_w<<<(HH * HH + 255) / 256, 256, 0, stream>>>(Watt1, Wat1T, HH, HH);

    k_similarity <<<BB * 32 * 8, 32, 0, stream>>>(cnb, adjb, rowsum, &scal[2]);
    k_diffuse    <<<BB * 32 * 4, 32, 0, stream>>>(adjb, ctxT, ctx, rowsum, hinb, &scal[0]);
    k_rowq       <<<BB * NN / 256, 256, 0, stream>>>(rowsum, sqnorm, &scal[1]);
    k_gcn        <<<BB * 32 * 4, 32, 0, stream>>>(hinb, WgcnT, bgcn, hbuf, hb);
    k_att1       <<<BB * 32 * 4, 32, 0, stream>>>(hb, Wat1T, batt1, abuf);
    k_lngelu     <<<BB * NN / 8, 256, 0, stream>>>(abuf, lng, lnb, Watt2, batt2, amask, wlog);
    k_pool       <<<BB, 256, 0, stream>>>(wlog, hbuf, Wout, bout, out);
    k_finalize   <<<1, 1, 0, stream>>>(scal, out);
}
